// MultiHeadAttention_75076028334624
// MI455X (gfx1250) — matrix operand order repaired and checked by reading, not compiled, not run on hardware
//
#include <hip/hip_runtime.h>

// MHA forward for MI455X (gfx1250, wave32, WMMA + Tensor Data Mover).
// SEQ=1024, BATCH=8, EMBED=768, HEADS=12, HDIM=64.
// Pipeline: W->bf16 transpose | bf16 WMMA projections (Q,K row-major, V transposed)
// | flash-attention: TDM double-buffered K/V^T tiles in LDS, online softmax,
//   all matmuls on v_wmma_f32_16x16x32_bf16.
// Workspace: 3*768*768 + 3*96*1024*64 bf16 elems ~= 39.4 MB of d_ws.

#define SEQ   1024
#define BATCH 8
#define EMBED 768
#define HEADS 12
#define HDIM  64
#define BHN   (BATCH * HEADS) // 96

typedef __bf16 bf16;
typedef __bf16 v16bf __attribute__((ext_vector_type(16)));
typedef __bf16 v8bf  __attribute__((ext_vector_type(8)));
typedef __bf16 v4bf  __attribute__((ext_vector_type(4)));
typedef float  v8f   __attribute__((ext_vector_type(8)));

#if __has_builtin(__builtin_amdgcn_tensor_load_to_lds) && \
    __has_builtin(__builtin_amdgcn_s_wait_tensorcnt)
#define USE_TDM 1
#else
#define USE_TDM 0
#endif

// ---- WMMA helpers -----------------------------------------------------------

__device__ __forceinline__ v8f bf16_mma(v16bf a, v16bf b, v8f c) {
  // D = A(16x32) * B(32x16) + C(16x16), fp32 accumulate
  return __builtin_amdgcn_wmma_f32_16x16x32_bf16(
      /*neg_a=*/false, a, /*neg_b=*/false, b,
      /*c_mod=*/(short)0, c, /*reuse_a=*/false, /*reuse_b=*/false);
}

// A fragment: lane holds row (M = lane%16); element e -> k = e + 16*half.
__device__ __forceinline__ v16bf a_frag(const bf16* __restrict__ row, int hlf) {
  return *(const v16bf*)(row + 16 * hlf);
}

// B fragment: lane holds column (N = lane%16); element e -> k = e + 16*half.
__device__ __forceinline__ v16bf b_frag(const bf16* __restrict__ col, int hlf) {
  return *(const v16bf*)(col + 16 * hlf);
}

// ---- TDM: 2D tile load (bf16, data_size=2) ---------------------------------
// D# per CDNA5 ISA ch.8: group0 {count=1, lds_addr, global_addr, type=2},
// group1 {data_size, tensor_dim0/1, tile_dim0/1, tensor_dim0_stride}.
// Groups 2/3 zero (<=2D tensor). Descriptors are workgroup-uniform -> SGPRs.

#if USE_TDM
typedef unsigned int u32x4 __attribute__((ext_vector_type(4)));
typedef int          i32x8 __attribute__((ext_vector_type(8)));
typedef int          i32x4 __attribute__((ext_vector_type(4)));

__device__ __forceinline__ void tdm_load_2d(const bf16* gsrc, const bf16* ldst,
                                            unsigned tile0, unsigned tile1,
                                            unsigned tdim0, unsigned tdim1,
                                            unsigned long long stride0) {
  unsigned long long ga = (unsigned long long)(size_t)(const void*)gsrc;
  unsigned lds_off = (unsigned)(size_t)(const void*)ldst; // low 32b = LDS offset
  u32x4 g0;
  g0[0] = 1u;                                         // count=1 valid descriptor
  g0[1] = lds_off;                                    // lds_addr
  g0[2] = (unsigned)(ga & 0xFFFFFFFFu);               // global_addr[31:0]
  g0[3] = (unsigned)((ga >> 32) & 0x1FFFFFFu) | (2u << 30); // addr[56:32]|type=2
  i32x8 g1;
  g1[0] = (int)(1u << 16);                            // data_size=1 -> 2 bytes
  g1[1] = (int)((tdim0 & 0xFFFFu) << 16);             // tensor_dim0[15:0]
  g1[2] = (int)((tdim0 >> 16) | ((tdim1 & 0xFFFFu) << 16));
  g1[3] = (int)((tdim1 >> 16) | (tile0 << 16));       // tile_dim0
  g1[4] = (int)(tile1 & 0xFFFFu);                     // tile_dim1, tile_dim2=0
  g1[5] = (int)(stride0 & 0xFFFFFFFFull);             // tensor_dim0_stride lo
  g1[6] = (int)((stride0 >> 32) & 0xFFFFull);         // stride0 hi, stride1=0
  g1[7] = 0;
  i32x4 z4 = {};
#if defined(__clang_major__) && (__clang_major__ >= 23)
  i32x8 z8 = {};
  __builtin_amdgcn_tensor_load_to_lds(g0, g1, z4, z4, z8, 0);
#else
  __builtin_amdgcn_tensor_load_to_lds(g0, g1, z4, z4, 0);
#endif
}
#endif // USE_TDM

// ---- Kernel 1: W [in][out] fp32 -> Wt [out][in] bf16 (x3 matrices) ----------

__global__ __launch_bounds__(256) void wprep_kernel(
    const float* __restrict__ Wq, const float* __restrict__ Wk,
    const float* __restrict__ Wv, bf16* __restrict__ Wt) {
  size_t idx = (size_t)blockIdx.x * 256 + threadIdx.x; // < 3*768*768
  int m   = (int)(idx / (EMBED * EMBED));
  int rem = (int)(idx % (EMBED * EMBED));
  int oc = rem / EMBED, ic = rem % EMBED;
  const float* W = (m == 0) ? Wq : (m == 1) ? Wk : Wv;
  Wt[idx] = (bf16)W[(size_t)ic * EMBED + oc];
}

// ---- Kernel 2: projection GEMM  Y = X @ W + b -------------------------------
// X: [8192 rows = s*8+b][768] fp32.  Wt: [768 out][768 in] bf16.
// vmode==0: dst[(b*12+h)*1024 + s][64] (Q/K);  vmode==1: dst[(b*12+h)*64 + d][1024] (V^T)

__global__ __launch_bounds__(128) void proj_kernel(
    const float* __restrict__ X, const bf16* __restrict__ Wt,
    const float* __restrict__ bias, bf16* __restrict__ dst, int vmode) {
  __shared__ __attribute__((aligned(32))) bf16 Xs[64 * 32];
  __shared__ __attribute__((aligned(32))) bf16 Ws[64 * 32];
  const int t = threadIdx.x;
  const int lane = t & 31, wave = t >> 5;
  const int hlf = lane >> 4, ln = lane & 15;
  const int r0 = blockIdx.x * 64; // output rows (s*B+b)
  const int c0 = blockIdx.y * 64; // output cols (h*64+d)

  v8f acc[4] = {};

  for (int k0 = 0; k0 < EMBED; k0 += 32) {
#pragma unroll
    for (int i = 0; i < 4; ++i) {
      int idx = t + i * 128;          // float4 units, 512 total
      int row = idx >> 3, c4 = idx & 7;
      float4 xv = *(const float4*)(X + (size_t)(r0 + row) * EMBED + k0 + c4 * 4);
      v4bf bv;
      bv[0] = (bf16)xv.x; bv[1] = (bf16)xv.y; bv[2] = (bf16)xv.z; bv[3] = (bf16)xv.w;
      *(v4bf*)(Xs + row * 32 + c4 * 4) = bv;
    }
#pragma unroll
    for (int i = 0; i < 4; ++i) {
      int idx = t + i * 128;          // v4bf units, 512 total
      int row = idx >> 3, c4 = idx & 7;
      *(v4bf*)(Ws + row * 32 + c4 * 4) =
          *(const v4bf*)(Wt + (size_t)(c0 + row) * EMBED + k0 + c4 * 4);
    }
    __syncthreads();

    v16bf a = a_frag(Xs + (wave * 16 + ln) * 32, hlf);
#pragma unroll
    for (int nb = 0; nb < 4; ++nb) {
      v16bf b = b_frag(Ws + (nb * 16 + ln) * 32, hlf);
      acc[nb] = bf16_mma(a, b, acc[nb]);
    }
    __syncthreads();
  }

#pragma unroll
  for (int nb = 0; nb < 4; ++nb) {
    int c = c0 + nb * 16 + ln;
    int h = c >> 6, d = c & 63;
    float bc = bias[c];
#pragma unroll
    for (int v = 0; v < 8; ++v) {
      int r = r0 + wave * 16 + v + 8 * hlf;
      int s = r >> 3, b = r & 7;
      int bh = b * HEADS + h;
      float val = acc[nb][v] + bc;
      size_t di = vmode ? ((size_t)(bh * 64 + d) * SEQ + s)
                        : (((size_t)bh * SEQ + s) * 64 + d);
      dst[di] = (bf16)val;
    }
  }
}

// ---- Kernel 3: flash attention ---------------------------------------------
// Grid: x = 16 query tiles of 64, y = 96 (b*H+h). Block = 128 (4 waves).
// K/V^T tiles double-buffered in LDS; TDM prefetches tile j+1 while WMMA runs
// on tile j. Wave 0 issues DMA + s_wait_tensorcnt; barrier publishes to the WG.

__global__ __launch_bounds__(128) void attn_kernel(
    const bf16* __restrict__ Q, const bf16* __restrict__ K,
    const bf16* __restrict__ Vt, float* __restrict__ Out) {
  __shared__ __attribute__((aligned(32))) bf16 KsBuf[2][32 * 64]; // [key][d]
  __shared__ __attribute__((aligned(32))) bf16 VsBuf[2][64 * 32]; // [d][key]
  __shared__ __attribute__((aligned(32))) bf16 Ps[4 * 16 * 32];   // per-wave P

  const int t = threadIdx.x;
  const int lane = t & 31, wave = t >> 5;
  const int hlf = lane >> 4, ln = lane & 15;
  const int bh = blockIdx.y;
  const int b = bh / HEADS, h = bh % HEADS;
  const int qBase = blockIdx.x * 64 + wave * 16;

  // Persistent Q fragments for this wave's 16 rows (d chunks 0-31, 32-63).
  const bf16* qrow = Q + ((size_t)bh * SEQ + qBase + ln) * 64;
  v16bf qa0 = a_frag(qrow, hlf);
  v16bf qa1 = a_frag(qrow + 32, hlf);

  float m[8], l[8];
  v8f o[4] = {};
#pragma unroll
  for (int v = 0; v < 8; ++v) { m[v] = -3.0e38f; l[v] = 0.0f; }

  const bf16* kbase = K + (size_t)bh * SEQ * 64;
  const bf16* vbase = Vt + (size_t)bh * 64 * SEQ;
  const float scale = 0.125f; // 1/sqrt(64)
  const int NBLK = SEQ / 32;  // 32 key blocks

#if USE_TDM
  if (wave == 0) {
    // Prologue: DMA first K tile (contiguous 32x64) and V^T tile (64 rows,
    // 32 elems, row stride 1024) into buffer 0.
    tdm_load_2d(kbase, KsBuf[0], 2048u, 1u, 2048u, 1u, 2048ull);
    tdm_load_2d(vbase, VsBuf[0], 32u, 64u, 32u, 64u, 1024ull);
  }
#endif

  for (int jb = 0; jb < NBLK; ++jb) {
    const int j0 = jb * 32;
    bf16* Ks = KsBuf[jb & 1];
    bf16* Vs = VsBuf[jb & 1];

#if USE_TDM
    if (wave == 0) __builtin_amdgcn_s_wait_tensorcnt(0); // current tiles landed
    __syncthreads();                                     // publish to all waves
    if (wave == 0 && jb + 1 < NBLK) {
      const int jn = (jb + 1) * 32;
      tdm_load_2d(kbase + (size_t)jn * 64, KsBuf[(jb + 1) & 1],
                  2048u, 1u, 2048u, 1u, 2048ull);
      tdm_load_2d(vbase + jn, VsBuf[(jb + 1) & 1],
                  32u, 64u, 32u, 64u, 1024ull);
    }
#else
    // Fallback: synchronous cooperative staging.
#pragma unroll
    for (int i = 0; i < 4; ++i) {
      int idx = t + i * 128;
      *(v4bf*)(Ks + idx * 4) = *(const v4bf*)(kbase + (size_t)j0 * 64 + idx * 4);
    }
#pragma unroll
    for (int i = 0; i < 4; ++i) {
      int idx = t + i * 128;
      int d = idx >> 3, c4 = idx & 7;
      *(v4bf*)(Vs + d * 32 + c4 * 4) =
          *(const v4bf*)(vbase + (size_t)d * SEQ + j0 + c4 * 4);
    }
    __syncthreads();
#endif

    // Scores S[16 q][32 keys] = Q . K^T  (4 WMMA)
    v8f s0 = {}, s1 = {};
    {
      v16bf bk0 = b_frag(Ks + ln * 64, hlf);
      s0 = bf16_mma(qa0, bk0, s0);
      bk0 = b_frag(Ks + ln * 64 + 32, hlf);
      s0 = bf16_mma(qa1, bk0, s0);
      v16bf bk1 = b_frag(Ks + (16 + ln) * 64, hlf);
      s1 = bf16_mma(qa0, bk1, s1);
      bk1 = b_frag(Ks + (16 + ln) * 64 + 32, hlf);
      s1 = bf16_mma(qa1, bk1, s1);
    }

    // Online softmax. Row M = v + 8*half lives in lanes [16*half .. +15];
    // xor-shuffles with masks 1/2/4/8 reduce exactly over one row.
    float corr[8];
#pragma unroll
    for (int v = 0; v < 8; ++v) {
      float a0 = scale * s0[v], a1 = scale * s1[v];
      float mv = fmaxf(a0, a1);
      mv = fmaxf(mv, __shfl_xor(mv, 1));
      mv = fmaxf(mv, __shfl_xor(mv, 2));
      mv = fmaxf(mv, __shfl_xor(mv, 4));
      mv = fmaxf(mv, __shfl_xor(mv, 8));
      float mnew = fmaxf(m[v], mv);
      float c = __expf(m[v] - mnew);
      float p0 = __expf(a0 - mnew);
      float p1 = __expf(a1 - mnew);
      float rs = p0 + p1;
      rs += __shfl_xor(rs, 1);
      rs += __shfl_xor(rs, 2);
      rs += __shfl_xor(rs, 4);
      rs += __shfl_xor(rs, 8);
      l[v] = l[v] * c + rs;
      m[v] = mnew;
      corr[v] = c;
      int ql = v + 8 * hlf;
      Ps[(wave * 16 + ql) * 32 + ln]      = (bf16)p0; // keys j0+ln
      Ps[(wave * 16 + ql) * 32 + 16 + ln] = (bf16)p1; // keys j0+16+ln
    }
#pragma unroll
    for (int nb = 0; nb < 4; ++nb)
#pragma unroll
      for (int v = 0; v < 8; ++v) o[nb][v] *= corr[v];

    // O += P . V  (P via per-wave LDS round trip C-layout -> A-layout; 4 WMMA)
    v16bf pa = a_frag(Ps + (wave * 16 + ln) * 32, hlf);
#pragma unroll
    for (int nb = 0; nb < 4; ++nb) {
      v16bf bv = b_frag(Vs + (nb * 16 + ln) * 32, hlf);
      o[nb] = bf16_mma(pa, bv, o[nb]);
    }
    __syncthreads(); // all waves done with this buffer before it is re-filled
  }

  // Epilogue: normalize and scatter to Out[s][b][h*64+d] fp32.
#pragma unroll
  for (int nb = 0; nb < 4; ++nb) {
    int d = nb * 16 + ln;
#pragma unroll
    for (int v = 0; v < 8; ++v) {
      int sq = qBase + v + 8 * hlf;
      Out[((size_t)sq * BATCH + b) * EMBED + h * 64 + d] = o[nb][v] / l[v];
    }
  }
}

// ---- Host launcher ----------------------------------------------------------

extern "C" void kernel_launch(void* const* d_in, const int* in_sizes, int n_in,
                              void* d_out, int out_size, void* d_ws, size_t ws_size,
                              hipStream_t stream) {
  (void)in_sizes; (void)n_in; (void)out_size; (void)ws_size;
  const float* q  = (const float*)d_in[0];
  const float* k  = (const float*)d_in[1];
  const float* v  = (const float*)d_in[2];
  const float* Wq = (const float*)d_in[3];
  const float* bq = (const float*)d_in[4];
  const float* Wk = (const float*)d_in[5];
  const float* bk = (const float*)d_in[6];
  const float* Wv = (const float*)d_in[7];
  const float* bv = (const float*)d_in[8];
  float* out = (float*)d_out;

  bf16* wt  = (bf16*)d_ws;                          // 3*768*768
  bf16* qb  = wt + (size_t)3 * EMBED * EMBED;       // 96*1024*64
  bf16* kb  = qb + (size_t)BHN * SEQ * 64;
  bf16* vtb = kb + (size_t)BHN * SEQ * 64;

  wprep_kernel<<<(3 * EMBED * EMBED) / 256, 256, 0, stream>>>(Wq, Wk, Wv, wt);

  dim3 pg(8192 / 64, EMBED / 64); // 128 x 12
  proj_kernel<<<pg, 128, 0, stream>>>(q, wt,                             bq, qb,  0);
  proj_kernel<<<pg, 128, 0, stream>>>(k, wt + (size_t)EMBED * EMBED,     bk, kb,  0);
  proj_kernel<<<pg, 128, 0, stream>>>(v, wt + (size_t)2 * EMBED * EMBED, bv, vtb, 1);

  dim3 ag(SEQ / 64, BHN); // 16 x 96
  attn_kernel<<<ag, 128, 0, stream>>>(qb, kb, vtb, out);
}
